// Genotype2PhenotypeTransformer_16303695855714
// MI455X (gfx1250) — matrix operand order repaired and checked by reading, not compiled, not run on hardware
//
#include <hip/hip_runtime.h>
#include <hip/hip_bf16.h>
#include <math.h>

// ---- problem constants (match reference) ----
#define B_   32
#define NS_  4096
#define D_   256
#define H_   4
#define L_   6
#define NQ_  512
#define NK_  1024
#define DH_  64
#define SCALE_ 0.125f   // 1/sqrt(64)

typedef __attribute__((ext_vector_type(16))) _Float16 v16h;
typedef __attribute__((ext_vector_type(8)))  _Float16 v8h;
typedef __attribute__((ext_vector_type(8)))  float    v8f;

static __device__ __forceinline__ v16h cat16(v8h a, v8h b) {
  v16h r;
#pragma unroll
  for (int i = 0; i < 8; ++i) { r[i] = a[i]; r[i + 8] = b[i]; }
  return r;
}

static __device__ __forceinline__ v8f wmma16(v16h a, v16h b, v8f c) {
  // D = A(16x32 f16) * B(32x16 f16) + C(16x16 f32)
  return __builtin_amdgcn_wmma_f32_16x16x32_f16(false, a, false, b, (short)0, c,
                                                false, false);
}

static __device__ __forceinline__ float gelu_tanh(float x) {
  float x3 = x * x * x;
  return 0.5f * x * (1.0f + tanhf(0.7978845608f * (x + 0.044715f * x3)));
}

// block = 256 threads, two values reduced simultaneously; sbuf needs 18 floats
static __device__ __forceinline__ void block_reduce2(float& a, float& b,
                                                     float* sbuf, int tid) {
#pragma unroll
  for (int off = 16; off > 0; off >>= 1) {
    a += __shfl_xor(a, off, 32);
    b += __shfl_xor(b, off, 32);
  }
  int wid = tid >> 5, lane = tid & 31;
  if (lane == 0) { sbuf[wid] = a; sbuf[8 + wid] = b; }
  __syncthreads();
  if (tid == 0) {
    float ta = 0.f, tb = 0.f;
#pragma unroll
    for (int i = 0; i < 8; ++i) { ta += sbuf[i]; tb += sbuf[8 + i]; }
    sbuf[16] = ta; sbuf[17] = tb;
  }
  __syncthreads();
  a = sbuf[16]; b = sbuf[17];
}

// ------------------------------------------------------------------ utilities
__global__ void zero_kernel(float* p, size_t n) {
  for (size_t i = (size_t)blockIdx.x * blockDim.x + threadIdx.x; i < n;
       i += (size_t)gridDim.x * blockDim.x)
    p[i] = 0.0f;
}

// weight convert + transpose: WT[c*256 + r] = (f16) W[r*256 + c]
__global__ void wcvt_t_kernel(const float* __restrict__ s,
                              _Float16* __restrict__ d) {
  int i = blockIdx.x * blockDim.x + threadIdx.x;  // i < 65536
  int r = i >> 8, c = i & 255;
  d[c * D_ + r] = (_Float16)s[i];
}

// V transpose: V[b,nk,D] (f16, head-sliced) -> VT[b,h,dh,NK] (f16)
// grid = (NK/64, H, B), block = 256; LDS 64x64 tile for coalescing both sides
__global__ void vtranspose_kernel(const _Float16* __restrict__ V,
                                  _Float16* __restrict__ VT) {
  __shared__ _Float16 tile[64 * 65];
  int t = threadIdx.x;
  int k0 = blockIdx.x * 64;
  int h = blockIdx.y, b = blockIdx.z;
#pragma unroll
  for (int rep = 0; rep < 16; ++rep) {
    int e = rep * 256 + t;
    int kr = e >> 6, dc = e & 63;
    tile[kr * 65 + dc] = V[((size_t)(b * NK_ + k0 + kr)) * D_ + h * DH_ + dc];
  }
  __syncthreads();
#pragma unroll
  for (int rep = 0; rep < 16; ++rep) {
    int e = rep * 256 + t;
    int dc = e >> 6, kr = e & 63;
    VT[(((size_t)(b * H_ + h)) * DH_ + dc) * NK_ + k0 + kr] = tile[kr * 65 + dc];
  }
}

// ------------------------------------------------- gather + sys-LayerNorm
// grid = (N, B), block = 256 (one thread per feature)
__global__ void gather_ln_kernel(const float* __restrict__ upd,
                                 const float* __restrict__ emb,
                                 const int* __restrict__ idx,
                                 const float* __restrict__ sc,
                                 const float* __restrict__ bi,
                                 _Float16* __restrict__ out, int N) {
  __shared__ float sbuf[18];
  int n = blockIdx.x, b = blockIdx.y, d = threadIdx.x;
  int row = idx[n];
  float x = emb[(size_t)row * D_ + d] + upd[((size_t)b * NS_ + row) * D_ + d];
  float s = x, s2 = x * x;
  block_reduce2(s, s2, sbuf, d);
  float m = s * (1.0f / D_);
  float var = s2 * (1.0f / D_) - m * m;
  float y = (x - m) * rsqrtf(var + 1e-5f) * sc[d] + bi[d];
  out[((size_t)b * N + n) * D_ + d] = (_Float16)y;
}

// ------------------------------------------------------------- WMMA GEMM
// C[rows,256] = act(A[rows,256] * W[256,256] + bias), W given PRE-TRANSPOSED
// (WT[n*256+k], f16, L2-resident). No LDS, no barriers; software-pipelined:
// fragments for k-step s+1 are loaded while the 4 WMMAs of step s execute.
// __launch_bounds__(256, 1): allow the full register budget (no spills);
// the pipeline supplies the latency hiding that occupancy would have.
// MODE 0: f16 out | 1: bias+gelu, f16 out | 2: bias, f32 out | 3: f32 out
// block = 256 (8 waves: 2 M-tiles x 4 N-tiles), grid.x = rows/32
template <int MODE>
__global__ __launch_bounds__(256, 1) void gemm256_kernel(
    const _Float16* __restrict__ A, const _Float16* __restrict__ WT,
    const float* __restrict__ bias, _Float16* __restrict__ C16,
    float* __restrict__ C32) {
  int tid  = threadIdx.x;
  int lane = tid & 31, w = tid >> 5;
  int hi = (lane >> 4) & 1, ln = lane & 15;
  int Mbase = blockIdx.x * 32 + (w & 1) * 16;
  int Nbase = (w >> 1) * 64;

  const _Float16* arow0 = A + ((size_t)(Mbase + ln)) * D_;

  auto loadA = [&](int s) -> v16h {
    const _Float16* p = arow0 + s * 32;
    return cat16(*(const v8h*)(p + 16 * hi), *(const v8h*)(p + 16 * hi + 8));
  };
  auto loadB = [&](int s, int c) -> v16h {
    const _Float16* p =
        WT + ((size_t)(Nbase + 16 * c + ln)) * D_ + s * 32 + 16 * hi;
    return cat16(*(const v8h*)p, *(const v8h*)(p + 8));
  };

  v8f acc[4] = {};
  v16h af = loadA(0);
  v16h bf[4];
#pragma unroll
  for (int c = 0; c < 4; ++c) bf[c] = loadB(0, c);

#pragma unroll
  for (int s = 0; s < 8; ++s) {
    v16h afn;
    v16h bfn[4];
    if (s < 7) {  // constant-folded under full unroll
      afn = loadA(s + 1);
#pragma unroll
      for (int c = 0; c < 4; ++c) bfn[c] = loadB(s + 1, c);
    }
#pragma unroll
    for (int c = 0; c < 4; ++c) acc[c] = wmma16(af, bf[c], acc[c]);
    if (s < 7) {
      af = afn;
#pragma unroll
      for (int c = 0; c < 4; ++c) bf[c] = bfn[c];
    }
  }

#pragma unroll
  for (int c = 0; c < 4; ++c) {
    int col = Nbase + 16 * c + ln;
    float bv = (MODE == 1 || MODE == 2) ? bias[col] : 0.0f;
#pragma unroll
    for (int r = 0; r < 8; ++r) {
      size_t row = (size_t)Mbase + r + 8 * hi;
      float v = acc[c][r] + bv;
      if (MODE == 1) v = gelu_tanh(v);
      if (MODE == 0 || MODE == 1)
        C16[row * D_ + col] = (_Float16)v;
      else
        C32[row * D_ + col] = v;
    }
  }
}

// --------------------------------------------------- flash attention (WMMA)
// block = 128 (4 waves, one head each); grid = (NQ/16, B)
// Software-pipelined ping-pong: chunk kb+1's K/V^T fragments (16 b128 loads)
// are in flight while chunk kb's WMMA + online-softmax VALU executes.
// __launch_bounds__(128, 1): the two fragment buffers + accumulators need
// ~240 VGPRs; one wave per SIMD (4 per WGP) with zero spills beats spilled
// higher occupancy.
// LDS only for the P-tile C-layout -> A-layout relayout (per-wave region,
// same-wave DS ops are in-order => no barrier).
__global__ __launch_bounds__(128, 1) void attention_kernel(
    const _Float16* __restrict__ Q, const _Float16* __restrict__ K,
    const _Float16* __restrict__ VT, const float* __restrict__ maskl,
    _Float16* __restrict__ ctx) {
  __shared__ __align__(16) _Float16 PT[H_][16 * 40];
  int tid = threadIdx.x;
  int w = tid >> 5, lane = tid & 31;
  int hi = (lane >> 4) & 1, ln = lane & 15;
  int h = w;
  int qb = blockIdx.x * 16, b = blockIdx.y;

  const _Float16* qrow = Q + ((size_t)(b * NQ_ + qb + ln)) * D_ + h * DH_;
  v16h aq0 = cat16(*(const v8h*)(qrow + 16 * hi),
                   *(const v8h*)(qrow + 16 * hi + 8));
  v16h aq1 = cat16(*(const v8h*)(qrow + 32 + 16 * hi),
                   *(const v8h*)(qrow + 32 + 16 * hi + 8));

  const _Float16* vt = VT + ((size_t)(b * H_ + h)) * DH_ * NK_;

  float m[8], ls[8];
#pragma unroll
  for (int r = 0; r < 8; ++r) { m[r] = -1e30f; ls[r] = 0.0f; }
  v8f acc[4] = {};

  auto loadChunk = [&](int kb, v16h (&bk)[2][2], v16h (&bv)[4]) {
    int k0 = kb * 32;
#pragma unroll
    for (int t = 0; t < 2; ++t) {
      const _Float16* krow =
          K + ((size_t)(b * NK_ + k0 + 16 * t + ln)) * D_ + h * DH_;
      __builtin_prefetch(krow + 64 * D_, 0, 1);
      bk[t][0] = cat16(*(const v8h*)(krow + 16 * hi),
                       *(const v8h*)(krow + 16 * hi + 8));
      bk[t][1] = cat16(*(const v8h*)(krow + 32 + 16 * hi),
                       *(const v8h*)(krow + 32 + 16 * hi + 8));
    }
#pragma unroll
    for (int c = 0; c < 4; ++c) {
      const _Float16* vrow = vt + ((size_t)(16 * c + ln)) * NK_ + k0 + 16 * hi;
      bv[c] = cat16(*(const v8h*)vrow, *(const v8h*)(vrow + 8));
    }
  };

  auto computeChunk = [&](int kb, v16h (&bk)[2][2], v16h (&bv)[4]) {
    int k0 = kb * 32;
    v8f zero = {};
    // scores: S(16x32) = Q(16x64) x K^T(64x32)
    v8f sc[2];
#pragma unroll
    for (int t = 0; t < 2; ++t) {
      sc[t] = wmma16(aq0, bk[t][0], zero);
      sc[t] = wmma16(aq1, bk[t][1], sc[t]);
    }
    // mask + scale (C layout: row = r + 8*hi, col = 16*t + ln)
#pragma unroll
    for (int t = 0; t < 2; ++t) {
#pragma unroll
      for (int r = 0; r < 8; ++r) {
        int q = qb + r + 8 * hi;
        int kc = k0 + 16 * t + ln;
        float mv = maskl[(size_t)q * NK_ + kc];
        sc[t][r] = (mv > 0.5f) ? sc[t][r] * SCALE_ : -1e9f;
      }
    }
    // online softmax over this 32-column chunk
#pragma unroll
    for (int r = 0; r < 8; ++r) {
      float rm = fmaxf(sc[0][r], sc[1][r]);
#pragma unroll
      for (int off = 1; off < 16; off <<= 1)
        rm = fmaxf(rm, __shfl_xor(rm, off, 32));
      float nm = fmaxf(m[r], rm);
      float alpha = __expf(m[r] - nm);
      float p0 = __expf(sc[0][r] - nm);
      float p1 = __expf(sc[1][r] - nm);
      float rs = p0 + p1;
#pragma unroll
      for (int off = 1; off < 16; off <<= 1) rs += __shfl_xor(rs, off, 32);
      ls[r] = ls[r] * alpha + rs;
      m[r] = nm;
#pragma unroll
      for (int c = 0; c < 4; ++c) acc[c][r] *= alpha;
      PT[w][(r + 8 * hi) * 40 + ln] = (_Float16)p0;
      PT[w][(r + 8 * hi) * 40 + 16 + ln] = (_Float16)p1;
    }
    // ctx += P(16x32) x V(32x64); same-wave DS is in-order
    v16h ap = cat16(*(const v8h*)&PT[w][ln * 40 + 16 * hi],
                    *(const v8h*)&PT[w][ln * 40 + 16 * hi + 8]);
#pragma unroll
    for (int c = 0; c < 4; ++c) acc[c] = wmma16(ap, bv[c], acc[c]);
  };

  v16h bkA[2][2], bvA[4], bkB[2][2], bvB[4];
  loadChunk(0, bkA, bvA);
  for (int kb = 0; kb < NK_ / 32; kb += 2) {
    loadChunk(kb + 1 < NK_ / 32 ? kb + 1 : kb, bkB, bvB);
    computeChunk(kb, bkA, bvA);
    loadChunk(kb + 2 < NK_ / 32 ? kb + 2 : kb, bkA, bvA);
    computeChunk(kb + 1, bkB, bvB);
  }

  // write ctx (f16)
#pragma unroll
  for (int c = 0; c < 4; ++c) {
#pragma unroll
    for (int r = 0; r < 8; ++r) {
      int row = qb + r + 8 * hi;
      int col = h * DH_ + 16 * c + ln;
      ctx[((size_t)b * NQ_ + row) * D_ + col] = (_Float16)(acc[c][r] / ls[r]);
    }
  }
}

// --------------------------------------------- inner LN: f32 in -> f32+f16 out
__global__ void ln_kernel(const float* __restrict__ in,
                          const float* __restrict__ sc,
                          const float* __restrict__ bi,
                          float* __restrict__ out32,
                          _Float16* __restrict__ out16) {
  __shared__ float sbuf[18];
  size_t row = blockIdx.x;
  int d = threadIdx.x;
  float x = in[row * D_ + d];
  float s = x, s2 = x * x;
  block_reduce2(s, s2, sbuf, d);
  float mm = s * (1.0f / D_);
  float var = s2 * (1.0f / D_) - mm * mm;
  float y = (x - mm) * rsqrtf(var + 1e-5f) * sc[d] + bi[d];
  out32[row * D_ + d] = y;
  out16[row * D_ + d] = (_Float16)y;
}

// ----------------- residual + outer LN + eff LN + atomic scatter into 2 bufs
__global__ void resid_scatter_kernel(const float* __restrict__ h,
                                     const float* __restrict__ f,
                                     const float* __restrict__ osc,
                                     const float* __restrict__ obi,
                                     const float* __restrict__ esc,
                                     const float* __restrict__ ebi,
                                     const int* __restrict__ qidx,
                                     float* __restrict__ upd,
                                     float* __restrict__ outr) {
  __shared__ float sbuf[18];
  int n = blockIdx.x, b = blockIdx.y, d = threadIdx.x;
  size_t row = (size_t)b * NQ_ + n;
  float x = h[row * D_ + d] + f[row * D_ + d];
  float s = x, s2 = x * x;
  block_reduce2(s, s2, sbuf, d);
  float mm = s * (1.0f / D_);
  float var = s2 * (1.0f / D_) - mm * mm;
  float y = (x - mm) * rsqrtf(var + 1e-5f) * osc[d] + obi[d];
  // second LN (eff)
  float t = y, t2 = y * y;
  block_reduce2(t, t2, sbuf, d);
  float m2 = t * (1.0f / D_);
  float v2 = t2 * (1.0f / D_) - m2 * m2;
  float z = (y - m2) * rsqrtf(v2 + 1e-5f) * esc[d] + ebi[d];
  int tgt = qidx[n];
  atomicAdd(&upd[((size_t)b * NS_ + tgt) * D_ + d], z);
  atomicAdd(&outr[((size_t)b * NS_ + tgt) * D_ + d], z);
}

// ---------------------------------------------------------------------------
extern "C" void kernel_launch(void* const* d_in, const int* in_sizes, int n_in,
                              void* d_out, int out_size, void* d_ws,
                              size_t ws_size, hipStream_t stream) {
  const float* upd_in = (const float*)d_in[0];
  const float* emb    = (const float*)d_in[1];
  const float* maskL  = (const float*)d_in[2];
  const float* Wq = (const float*)d_in[3];
  const float* Wk = (const float*)d_in[4];
  const float* Wv = (const float*)d_in[5];
  const float* Wo = (const float*)d_in[6];
  const float* W1 = (const float*)d_in[7];
  const float* b1 = (const float*)d_in[8];
  const float* W2 = (const float*)d_in[9];
  const float* b2 = (const float*)d_in[10];
  const float* sys_s = (const float*)d_in[11];
  const float* sys_b = (const float*)d_in[12];
  const float* eff_s = (const float*)d_in[13];
  const float* eff_b = (const float*)d_in[14];
  const float* inn_s = (const float*)d_in[15];
  const float* inn_b = (const float*)d_in[16];
  const float* out_s = (const float*)d_in[17];
  const float* out_b = (const float*)d_in[18];
  const int* qidx = (const int*)d_in[19];
  const int* kidx = (const int*)d_in[20];
  float* out = (float*)d_out;

  // ---- workspace layout ----
  char* ws = (char*)d_ws;
  size_t off = 0;
  auto alloc = [&](size_t bytes) -> void* {
    void* p = ws + off;
    off += (bytes + 255) & ~(size_t)255;
    return p;
  };
  const size_t nUpd = (size_t)B_ * NS_ * D_;
  float*    upd   = (float*)alloc(nUpd * 4);
  _Float16* wq16  = (_Float16*)alloc((size_t)D_ * D_ * 2);  // all transposed
  _Float16* wk16  = (_Float16*)alloc((size_t)D_ * D_ * 2);
  _Float16* wv16  = (_Float16*)alloc((size_t)D_ * D_ * 2);
  _Float16* wo16  = (_Float16*)alloc((size_t)D_ * D_ * 2);
  _Float16* w116  = (_Float16*)alloc((size_t)D_ * D_ * 2);
  _Float16* w216  = (_Float16*)alloc((size_t)D_ * D_ * 2);
  _Float16* q16   = (_Float16*)alloc((size_t)B_ * NQ_ * D_ * 2);
  _Float16* k16   = (_Float16*)alloc((size_t)B_ * NK_ * D_ * 2);
  _Float16* Q16   = (_Float16*)alloc((size_t)B_ * NQ_ * D_ * 2);
  _Float16* K16   = (_Float16*)alloc((size_t)B_ * NK_ * D_ * 2);
  _Float16* V16   = (_Float16*)alloc((size_t)B_ * NK_ * D_ * 2);
  _Float16* VTp   = (_Float16*)alloc((size_t)B_ * NK_ * D_ * 2);  // [b,h,dh,NK]
  _Float16* ctx16 = (_Float16*)alloc((size_t)B_ * NQ_ * D_ * 2);
  _Float16* h16   = (_Float16*)alloc((size_t)B_ * NQ_ * D_ * 2);
  _Float16* t116  = (_Float16*)alloc((size_t)B_ * NQ_ * D_ * 2);
  float*    t232  = (float*)alloc((size_t)B_ * NQ_ * D_ * 4);
  float*    h32   = (float*)alloc((size_t)B_ * NQ_ * D_ * 4);

  // ---- init: working copy of update_tensor, zero output, f16 W^T ----
  hipMemcpyAsync(upd, upd_in, nUpd * 4, hipMemcpyDeviceToDevice, stream);
  zero_kernel<<<4096, 256, 0, stream>>>(out, nUpd);
  const int nW = D_ * D_;
  wcvt_t_kernel<<<nW / 256, 256, 0, stream>>>(Wq, wq16);
  wcvt_t_kernel<<<nW / 256, 256, 0, stream>>>(Wk, wk16);
  wcvt_t_kernel<<<nW / 256, 256, 0, stream>>>(Wv, wv16);
  wcvt_t_kernel<<<nW / 256, 256, 0, stream>>>(Wo, wo16);
  wcvt_t_kernel<<<nW / 256, 256, 0, stream>>>(W1, w116);
  wcvt_t_kernel<<<nW / 256, 256, 0, stream>>>(W2, w216);

  const int rowsQ = B_ * NQ_;   // 16384
  const int rowsK = B_ * NK_;   // 32768

  for (int l = 0; l < L_; ++l) {
    const int* qi = qidx + (size_t)l * NQ_;
    const int* ki = kidx + (size_t)l * NK_;
    const float* ml = maskL + (size_t)l * NQ_ * NK_;

    gather_ln_kernel<<<dim3(NQ_, B_), 256, 0, stream>>>(upd, emb, qi, sys_s,
                                                        sys_b, q16, NQ_);
    gather_ln_kernel<<<dim3(NK_, B_), 256, 0, stream>>>(upd, emb, ki, sys_s,
                                                        sys_b, k16, NK_);

    gemm256_kernel<0><<<rowsQ / 32, 256, 0, stream>>>(q16, wq16, nullptr, Q16,
                                                      nullptr);
    gemm256_kernel<0><<<rowsK / 32, 256, 0, stream>>>(k16, wk16, nullptr, K16,
                                                      nullptr);
    gemm256_kernel<0><<<rowsK / 32, 256, 0, stream>>>(k16, wv16, nullptr, V16,
                                                      nullptr);
    vtranspose_kernel<<<dim3(NK_ / 64, H_, B_), 256, 0, stream>>>(V16, VTp);

    attention_kernel<<<dim3(NQ_ / 16, B_), 128, 0, stream>>>(Q16, K16, VTp, ml,
                                                             ctx16);

    gemm256_kernel<3><<<rowsQ / 32, 256, 0, stream>>>(ctx16, wo16, nullptr,
                                                      nullptr, t232);
    ln_kernel<<<rowsQ, 256, 0, stream>>>(t232, inn_s, inn_b, h32, h16);

    gemm256_kernel<1><<<rowsQ / 32, 256, 0, stream>>>(h16, w116, b1, t116,
                                                      nullptr);
    gemm256_kernel<2><<<rowsQ / 32, 256, 0, stream>>>(t116, w216, b2, nullptr,
                                                      t232);

    resid_scatter_kernel<<<dim3(NQ_, B_), 256, 0, stream>>>(
        h32, t232, out_s, out_b, eff_s, eff_b, qi, upd, out);
  }
  (void)in_sizes; (void)n_in; (void)out_size; (void)ws_size;
}
